// MultiHeadAttention_62577673503287
// MI455X (gfx1250) — matrix operand order repaired and checked by reading, not compiled, not run on hardware
//
#include <hip/hip_runtime.h>
#include <hip/hip_bf16.h>
#include <math.h>

// Problem dims (compile-time)
#define Bz   4
#define Nn   2048
#define Ee   512
#define Hh   8
#define HDd  64
#define F3   1536
#define SCALE 0.125f   // HD^-0.5 = 64^-0.5

typedef __attribute__((ext_vector_type(16))) _Float16 v16h;
typedef __attribute__((ext_vector_type(8)))  _Float16 v8h;
typedef __attribute__((ext_vector_type(4)))  _Float16 v4h;
typedef __attribute__((ext_vector_type(8)))  float    v8f;
typedef __attribute__((ext_vector_type(4)))  float    v4f;

__device__ __forceinline__ v8h ld8h(const _Float16* p) {
    return *(const v8h*)p;
}

__device__ __forceinline__ v16h cat16(v8h lo, v8h hi) {
    return __builtin_shufflevector(lo, hi, 0,1,2,3,4,5,6,7,8,9,10,11,12,13,14,15);
}

__device__ __forceinline__ v8f wmma_f16(v16h a, v16h b, v8f c) {
    // D(16x16,f32) = A(16x32,f16) * B(32x16,f16) + C
    return __builtin_amdgcn_wmma_f32_16x16x32_f16(
        /*neg_a=*/false, a, /*neg_b=*/false, b,
        /*c_mod=*/(short)0, c, /*reuse_a=*/false, /*reuse_b=*/false);
}

// -----------------------------------------------------------------------------
// Kernel 0: one-shot f32 -> f16 conversion (x, w_qkv, w_out).  n % 4 == 0.
// -----------------------------------------------------------------------------
__global__ __launch_bounds__(256) void cvt_kernel(
    const float* __restrict__ src, _Float16* __restrict__ dst, int n4)
{
    const int i = blockIdx.x * 256 + threadIdx.x;
    if (i < n4) {
        v4f a = *(const v4f*)(src + (size_t)i * 4);
        v4h r;
#pragma unroll
        for (int j = 0; j < 4; ++j) r[j] = (_Float16)a[j];
        *(v4h*)(dst + (size_t)i * 4) = r;
    }
}

// -----------------------------------------------------------------------------
// Kernel 1: QKV projection.  qkv[row,f] = sum_e xh[row,e]*wh[f,e] + bias[f]
// One wave per 16x64 output tile.  Writes:
//   Q (pre-scaled) -> [B,H,N,64] f16
//   K              -> [B,H,N,64] f16
//   V (transposed) -> [B,H,64,N] f16
// -----------------------------------------------------------------------------
__global__ __launch_bounds__(32) void qkv_kernel(
    const _Float16* __restrict__ xh, const _Float16* __restrict__ wh,
    const float* __restrict__ bias,
    _Float16* __restrict__ q, _Float16* __restrict__ k, _Float16* __restrict__ v)
{
    const int lane = threadIdx.x;
    const int ln = lane & 15;
    const int hi = lane >> 4;
    const int row0 = blockIdx.x * 16;   // 0..8191 step 16
    const int f0   = blockIdx.y * 64;   // 0..1535 step 64

    v8f acc[4] = {};
    const _Float16* xrow = xh + (size_t)(row0 + ln) * Ee;

    for (int e0 = 0; e0 < Ee; e0 += 32) {
        // A tile: row m=ln, K contiguous {0-15}/{16-31}
        v16h A = cat16(ld8h(xrow + e0 + hi * 16),
                       ld8h(xrow + e0 + hi * 16 + 8));
#pragma unroll
        for (int g = 0; g < 4; ++g) {
            const _Float16* wp = wh + (size_t)(f0 + g * 16 + ln) * Ee + e0 + hi * 16;
            __builtin_prefetch(wp + 32, 0, 0);
            v16h Bm = cat16(ld8h(wp), ld8h(wp + 8));
            acc[g] = wmma_f16(A, Bm, acc[g]);
        }
    }

    // which tensor / head are uniform per wave (f0 multiple of 64)
    const int which = f0 >> 9;          // 0=q 1=k 2=v
    const int h     = (f0 >> 6) & 7;

#pragma unroll
    for (int g = 0; g < 4; ++g) {
        const int f = f0 + g * 16 + ln;
        const float bv = bias[f];
        const int d = f & 63;
#pragma unroll
        for (int r = 0; r < 8; ++r) {
            float val = acc[g][r] + bv;
            const int m = r + hi * 8;
            const int row = row0 + m;
            const int b = row >> 11;          // / 2048
            const int n = row & (Nn - 1);
            const int bh = b * Hh + h;
            if (which == 0)
                q[((size_t)bh * Nn + n) * HDd + d] = (_Float16)(val * SCALE);
            else if (which == 1)
                k[((size_t)bh * Nn + n) * HDd + d] = (_Float16)val;
            else
                v[((size_t)bh * HDd + d) * Nn + n] = (_Float16)val;
        }
    }
}

// -----------------------------------------------------------------------------
// Kernel 2: flash attention.  One wave per (b,h, 32 query rows).
// Online softmax over 32-key blocks; P goes through LDS to reach A-layout.
// K/V B-tiles are loaded once per key block and reused by both 16-row groups.
// -----------------------------------------------------------------------------
__global__ __launch_bounds__(32) void attn_kernel(
    const _Float16* __restrict__ q, const _Float16* __restrict__ k,
    const _Float16* __restrict__ v, _Float16* __restrict__ aout)
{
    __shared__ alignas(16) _Float16 pbuf[32 * 32];

    const int lane = threadIdx.x;
    const int ln = lane & 15;
    const int hi = lane >> 4;
    const int m0 = blockIdx.x * 32;     // query row tile (32 rows)
    const int bh = blockIdx.y;          // 0..31
    const int b  = bh >> 3;
    const int h  = bh & 7;

    const _Float16* qb = q + (size_t)bh * Nn * HDd;
    const _Float16* kb = k + (size_t)bh * Nn * HDd;
    const _Float16* vb = v + (size_t)bh * HDd * Nn;

    // Q A-tiles (Q is pre-scaled): [row group][feature half 0..31 / 32..63]
    v16h Aq[2][2];
#pragma unroll
    for (int rg = 0; rg < 2; ++rg) {
        const _Float16* qp = qb + (size_t)(m0 + rg * 16 + ln) * HDd;
        Aq[rg][0] = cat16(ld8h(qp + hi * 16),      ld8h(qp + hi * 16 + 8));
        Aq[rg][1] = cat16(ld8h(qp + 32 + hi * 16), ld8h(qp + 40 + hi * 16));
    }

    v8f O[2][4] = {};
    float mrow[2][8], lrow[2][8];
#pragma unroll
    for (int rg = 0; rg < 2; ++rg)
#pragma unroll
        for (int r = 0; r < 8; ++r) { mrow[rg][r] = -INFINITY; lrow[rg][r] = 0.0f; }

    for (int jb = 0; jb < Nn; jb += 32) {
        // ---- S = Q @ K^T for keys [jb, jb+32): 2 row groups x 2 key sub-tiles
        v8f S[2][2] = {};
#pragma unroll
        for (int t = 0; t < 2; ++t) {
            const _Float16* kp = kb + (size_t)(jb + t * 16 + ln) * HDd + hi * 16;
            v16h B0 = cat16(ld8h(kp),      ld8h(kp + 8));        // feat 0..31
            v16h B1 = cat16(ld8h(kp + 32), ld8h(kp + 40));       // feat 32..63
#pragma unroll
            for (int rg = 0; rg < 2; ++rg) {
                S[rg][t] = wmma_f16(Aq[rg][0], B0, S[rg][t]);
                S[rg][t] = wmma_f16(Aq[rg][1], B1, S[rg][t]);
            }
        }

        // ---- online softmax per row group ----
#pragma unroll
        for (int rg = 0; rg < 2; ++rg) {
            float alpha[8];
#pragma unroll
            for (int r = 0; r < 8; ++r) {
                float s = fmaxf(S[rg][0][r], S[rg][1][r]);
                s = fmaxf(s, __shfl_xor(s, 1, 32));
                s = fmaxf(s, __shfl_xor(s, 2, 32));
                s = fmaxf(s, __shfl_xor(s, 4, 32));
                s = fmaxf(s, __shfl_xor(s, 8, 32));
                const float mnew = fmaxf(mrow[rg][r], s);
                alpha[r] = __expf(mrow[rg][r] - mnew);
                mrow[rg][r] = mnew;
            }
#pragma unroll
            for (int r = 0; r < 8; ++r) {
                const float p0 = __expf(S[rg][0][r] - mrow[rg][r]);
                const float p1 = __expf(S[rg][1][r] - mrow[rg][r]);
                lrow[rg][r] = lrow[rg][r] * alpha[r] + p0 + p1;
                const int m = rg * 16 + r + hi * 8;
                pbuf[m * 32 + ln]      = (_Float16)p0;
                pbuf[m * 32 + 16 + ln] = (_Float16)p1;
#pragma unroll
                for (int g = 0; g < 4; ++g) O[rg][g][r] *= alpha[r];
            }
        }
        __syncthreads();

        // P in A-matrix layout (row=ln, K contiguous over 32 keys)
        v16h P[2];
#pragma unroll
        for (int rg = 0; rg < 2; ++rg)
            P[rg] = cat16(*(const v8h*)&pbuf[(rg * 16 + ln) * 32 + hi * 16],
                          *(const v8h*)&pbuf[(rg * 16 + ln) * 32 + hi * 16 + 8]);

        // ---- O += P @ V : V B-tile shared by both row groups ----
#pragma unroll
        for (int g = 0; g < 4; ++g) {
            const _Float16* vp = vb + (size_t)(g * 16 + ln) * Nn + jb + hi * 16;
            v16h Bv = cat16(ld8h(vp), ld8h(vp + 8));
            O[0][g] = wmma_f16(P[0], Bv, O[0][g]);
            O[1][g] = wmma_f16(P[1], Bv, O[1][g]);
        }
        __syncthreads();
    }

    // ---- normalize and store attention output (f16, [B*N, E]) ----
    const int e_base = h * HDd;
#pragma unroll
    for (int rg = 0; rg < 2; ++rg) {
#pragma unroll
        for (int r = 0; r < 8; ++r) {
            float l = lrow[rg][r];
            l += __shfl_xor(l, 1, 32);
            l += __shfl_xor(l, 2, 32);
            l += __shfl_xor(l, 4, 32);
            l += __shfl_xor(l, 8, 32);
            const float inv = 1.0f / l;
            const int m = rg * 16 + r + hi * 8;
            const size_t row = (size_t)b * Nn + (m0 + m);
#pragma unroll
            for (int g = 0; g < 4; ++g)
                aout[row * Ee + e_base + g * 16 + ln] = (_Float16)(O[rg][g][r] * inv);
        }
    }
}

// -----------------------------------------------------------------------------
// Kernel 3: output projection.  out[row,f] = sum_e a[row,e]*wh[f,e] + b[f]
// -----------------------------------------------------------------------------
__global__ __launch_bounds__(32) void proj_kernel(
    const _Float16* __restrict__ a, const _Float16* __restrict__ wh,
    const float* __restrict__ bias, float* __restrict__ out)
{
    const int lane = threadIdx.x;
    const int ln = lane & 15;
    const int hi = lane >> 4;
    const int row0 = blockIdx.x * 16;
    const int f0   = blockIdx.y * 64;

    v8f acc[4] = {};
    const _Float16* arow = a + (size_t)(row0 + ln) * Ee;

    for (int e0 = 0; e0 < Ee; e0 += 32) {
        v16h A = cat16(ld8h(arow + e0 + hi * 16),
                       ld8h(arow + e0 + hi * 16 + 8));
#pragma unroll
        for (int g = 0; g < 4; ++g) {
            const _Float16* wp = wh + (size_t)(f0 + g * 16 + ln) * Ee + e0 + hi * 16;
            __builtin_prefetch(wp + 32, 0, 0);
            v16h Bm = cat16(ld8h(wp), ld8h(wp + 8));
            acc[g] = wmma_f16(A, Bm, acc[g]);
        }
    }

#pragma unroll
    for (int g = 0; g < 4; ++g) {
        const int f = f0 + g * 16 + ln;
        const float bv = bias[f];
#pragma unroll
        for (int r = 0; r < 8; ++r)
            out[(size_t)(row0 + r + hi * 8) * Ee + f] = acc[g][r] + bv;
    }
}

// -----------------------------------------------------------------------------
extern "C" void kernel_launch(void* const* d_in, const int* in_sizes, int n_in,
                              void* d_out, int out_size, void* d_ws, size_t ws_size,
                              hipStream_t stream) {
    const float* x     = (const float*)d_in[0];
    const float* w_qkv = (const float*)d_in[1];
    const float* b_qkv = (const float*)d_in[2];
    const float* w_out = (const float*)d_in[3];
    const float* b_out = (const float*)d_in[4];
    float* out = (float*)d_out;

    // Workspace partition (f16 elements)
    _Float16* ws   = (_Float16*)d_ws;
    _Float16* q    = ws;                          //  4,194,304  (B*H*N*HD)
    _Float16* k    = ws + (size_t) 4194304;       //  4,194,304
    _Float16* v    = ws + (size_t) 8388608;       //  4,194,304
    _Float16* ao   = ws + (size_t)12582912;       //  4,194,304  (B*N*E)
    _Float16* xh   = ws + (size_t)16777216;       //  4,194,304  (B*N*E)
    _Float16* wqh  = ws + (size_t)20971520;       //    786,432  (3E*E)
    _Float16* woh  = ws + (size_t)21757952;       //    262,144  (E*E)
    // total: 22,020,096 halves = 42 MiB

    dim3 blk32(32);
    dim3 blk256(256);

    // one-shot f32 -> f16 conversions
    cvt_kernel<<<dim3(4194304 / 4 / 256), blk256, 0, stream>>>(x,     xh,  4194304 / 4);
    cvt_kernel<<<dim3( 786432 / 4 / 256), blk256, 0, stream>>>(w_qkv, wqh,  786432 / 4);
    cvt_kernel<<<dim3( 262144 / 4 / 256), blk256, 0, stream>>>(w_out, woh,  262144 / 4);

    // 8192/16 = 512 row tiles; 1536/64 = 24 col groups
    qkv_kernel<<<dim3(512, 24), blk32, 0, stream>>>(xh, wqh, b_qkv, q, k, v);
    // 2048/32 = 64 query tiles x 32 (b,h)
    attn_kernel<<<dim3(64, 32), blk32, 0, stream>>>(q, k, v, ao);
    // 512 row tiles x 512/64 = 8 col groups
    proj_kernel<<<dim3(512, 8), blk32, 0, stream>>>(ao, w_out ? woh : woh, b_out, out);
}
